// GraphAttentionLayer_77584289234977
// MI455X (gfx1250) — matrix operand order repaired and checked by reading, not compiled, not run on hardware
//
#include <hip/hip_runtime.h>

// GAT layer: B=64, N=307 (2N=614), D_IN=D_OUT=64.
// out[b,n,o] = ELU( sum_m P[n,m] * Wh[m,o]*rdenom[m] ), softmax-over-axis-1 factored
// into exact per-column (colmax, 1/denom) stats => no 96MB att tensor.

#define GAT_ALPHA 0.2f
#define NB   64
#define NN   307
#define N2   614
#define DD   64
#define RT1  10      // ceil(614/64) row tiles for Wh gemm
#define RT4  5       // ceil(614/128) row tiles for attention gemm

typedef __attribute__((ext_vector_type(16))) _Float16 v16h;
typedef __attribute__((ext_vector_type(8)))  float    v8f;

union FragU { v16h v; unsigned int u[8]; };

// A fragment (16x32 f16, ISA layout): lane<16 -> row=lane, K in {0..7,16..23};
// lane>=16 -> row=lane-16, K in {8..15,24..31}. rowptr: 32 halves (rows padded to 40).
__device__ __forceinline__ v16h load_fragA(const _Float16* rowptr, int lane) {
  const unsigned int* p = (const unsigned int*)rowptr;
  int b = (lane & 16) ? 4 : 0;
  FragU f;
#pragma unroll
  for (int i = 0; i < 4; ++i) f.u[i] = p[b + i];
#pragma unroll
  for (int i = 0; i < 4; ++i) f.u[4 + i] = p[b + 8 + i];
  return f.v;
}

// B fragment (32x16 f16): lane<16 -> col=lane, K in {0..7,16..23}; lane>=16 -> col=lane-16, K in {8..15,24..31}.
__device__ __forceinline__ v16h load_fragB(const _Float16* rowptr, int lane) {
  const unsigned int* p = (const unsigned int*)rowptr;
  int b = (lane & 16) ? 4 : 0;
  FragU f;
#pragma unroll
  for (int i = 0; i < 4; ++i) f.u[i] = p[b + i];
#pragma unroll
  for (int i = 0; i < 4; ++i) f.u[4 + i] = p[b + 8 + i];
  return f.v;
}

// ---------------- Kernel 1: Wh = concat(ht,h) @ W  (WMMA, f32 accum) ----------
__global__ __launch_bounds__(256) void k_wh_gemm(const float* __restrict__ h,
                                                 const float* __restrict__ ht,
                                                 const float* __restrict__ W,
                                                 float* __restrict__ Wh) {
  __shared__ _Float16 xt[64 * 40];   // x tile: [row][K]
  __shared__ _Float16 wt[64 * 40];   // W^T:    [col][K]
  int b = blockIdx.x / RT1, rt = blockIdx.x % RT1;
  int row0 = rt * 64;
  int t = threadIdx.x;
  int lane = t & 31, wave = t >> 5;
  int wrow = (wave & 3) * 16, wcol = (wave >> 2) * 32;
  v8f acc0 = {}, acc1 = {};

  for (int k0 = 0; k0 < 64; k0 += 32) {
    __syncthreads();
    {   // stage x tile (concat fused, branchless): 64 rows x 32 chans
      int tm = t & 31, tr = t >> 5;
#pragma unroll
      for (int r = 0; r < 8; ++r) {
        int rl = tr * 8 + r;
        int gr = row0 + rl;
        int gc = (gr < N2) ? gr : (N2 - 1);        // clamp -> always-valid addr
        const float* src = (gc < NN) ? &ht[(b * NN + gc) * DD]
                                     : &h[(b * NN + (gc - NN)) * DD];
        float v = src[k0 + tm];                     // unconditional load
        if (gr >= N2) v = 0.f;                      // mask value, not the load
        xt[rl * 40 + tm] = (_Float16)v;
      }
    }
    {   // stage W^T tile: wt[c][k] = W[k0+k][c]
      int c = t & 63, kb = t >> 6;
#pragma unroll
      for (int r = 0; r < 8; ++r) {
        int k = kb * 8 + r;
        wt[c * 40 + k] = (_Float16)W[(k0 + k) * DD + c];
      }
    }
    __syncthreads();
    v16h af = load_fragA(&xt[(wrow + (lane & 15)) * 40], lane);
    v16h b0 = load_fragB(&wt[(wcol + (lane & 15)) * 40], lane);
    v16h b1 = load_fragB(&wt[(wcol + 16 + (lane & 15)) * 40], lane);
    acc0 = __builtin_amdgcn_wmma_f32_16x16x32_f16(false, af, false, b0, (short)0, acc0, false, false);
    acc1 = __builtin_amdgcn_wmma_f32_16x16x32_f16(false, af, false, b1, (short)0, acc1, false, false);
  }

  int cM = (lane >> 4) * 8, cN = lane & 15;
#pragma unroll
  for (int g = 0; g < 8; ++g) {
    int n = row0 + wrow + cM + g;
    if (n < N2) {
      Wh[(b * N2 + n) * DD + wcol + cN]      = acc0[g];
      Wh[(b * N2 + n) * DD + wcol + 16 + cN] = acc1[g];
    }
  }
}

// ---------------- Kernel 2: Wh1/Wh2 = Wh . a[:64] / a[64:] ---------------------
__global__ void k_wh12(const float* __restrict__ Wh, const float* __restrict__ a,
                       float* __restrict__ wh1, float* __restrict__ wh2) {
  int idx = blockIdx.x * blockDim.x + threadIdx.x;
  if (idx >= NB * N2) return;
  const float4* row = (const float4*)&Wh[idx * DD];
  const float4* a1  = (const float4*)a;
  const float4* a2  = (const float4*)&a[DD];
  float s1 = 0.f, s2 = 0.f;
#pragma unroll
  for (int i = 0; i < 16; ++i) {
    float4 r4 = row[i], x1 = a1[i], x2 = a2[i];
    s1 += r4.x * x1.x + r4.y * x1.y + r4.z * x1.z + r4.w * x1.w;
    s2 += r4.x * x2.x + r4.y * x2.y + r4.z * x2.z + r4.w * x2.w;
  }
  wh1[idx] = s1;
  wh2[idx] = s2;
}

// -------- Kernel 3: per-column masked softmax stats (max + 1/denom, axis 1) ----
__global__ __launch_bounds__(256) void k_colstats(const float* __restrict__ adj,
                                                  const float* __restrict__ wh1,
                                                  const float* __restrict__ wh2,
                                                  float* __restrict__ cmax,
                                                  float* __restrict__ rdenom) {
  __shared__ float wh1s[N2];
  int b = blockIdx.x / 3, strip = blockIdx.x % 3;
  for (int i = threadIdx.x; i < N2; i += 256) wh1s[i] = wh1[b * N2 + i];
  __syncthreads();
  int j = strip * 256 + threadIdx.x;
  if (j >= N2) return;
  float w2 = wh2[b * N2 + j];
  float mx = -3.0e38f, s = 0.f;
  for (int i = 0; i < N2; ++i) {          // adj[i*N2+j]: coalesced across threads
    if (adj[i * N2 + j] > 0.f) {          // data-dependent: skips ~95% of exps
      float v = wh1s[i] + w2;
      v = v > 0.f ? v : GAT_ALPHA * v;
      if (v > mx) { s = s * __expf(mx - v) + 1.f; mx = v; }
      else        { s += __expf(v - mx); }
    }
  }
  cmax[b * N2 + j]   = mx;
  rdenom[b * N2 + j] = 1.0f / s;          // every column has >=1 unmasked entry
}

// -------- Kernel 4: out = ELU( P @ (Wh*rdenom) ) + concat-halves reshuffle -----
// Block tile: 128 rows x 64 cols; 8 waves, each wave 32x32 => 4 WMMA per K-chunk.
__global__ __launch_bounds__(256) void k_attn(const float* __restrict__ adj,
                                              const float* __restrict__ Wh,
                                              const float* __restrict__ wh1,
                                              const float* __restrict__ wh2,
                                              const float* __restrict__ cmax,
                                              const float* __restrict__ rdenom,
                                              float* __restrict__ out) {
  __shared__ _Float16 pt[128 * 40];  // P tile: [row][K=32]
  __shared__ _Float16 vt[64 * 40];   // V' transposed: [col][K=32]
  __shared__ float    wh1s[128];     // row logits, staged once per block
  int b = blockIdx.x / RT4, rt = blockIdx.x % RT4;
  int row0 = rt * 128;
  int t = threadIdx.x;
  int lane = t & 31, wave = t >> 5;
  int wrow = (wave & 3) * 32, wcol = (wave >> 2) * 32;
  v8f acc00 = {}, acc01 = {}, acc10 = {}, acc11 = {};

  if (t < 128) {
    int gr = row0 + t;
    wh1s[t] = (gr < N2) ? wh1[b * N2 + gr] : 0.f;
  }

  for (int k0 = 0; k0 < 640; k0 += 32) {   // 20 K-chunks over m (padded to 640)
    __syncthreads();
    {   // stage P tile (branchless): exp(lrelu(Wh1[n]+Wh2[m]) - colmax[m]) * mask
      int tm = t & 31, tr = t >> 5;
      int gm  = k0 + tm;
      bool mok = (gm < N2);
      int gmc = mok ? gm : (N2 - 1);                 // clamped column
      float w2 = wh2[b * N2 + gmc];                  // unconditional loads
      float cm = cmax[b * N2 + gmc];
#pragma unroll
      for (int r = 0; r < 16; ++r) {
        int rl = tr * 16 + r;
        int gr = row0 + rl;
        int grc = (gr < N2) ? gr : (N2 - 1);         // clamped row
        float adjv = adj[grc * N2 + gmc];            // unconditional, coalesced in tm
        float v = wh1s[rl] + w2;
        v = v > 0.f ? v : GAT_ALPHA * v;
        float e = __expf(v - cm);
        float pv = (adjv > 0.f && gr < N2 && mok) ? e : 0.f;
        pt[rl * 40 + tm] = (_Float16)pv;
      }
    }
    {   // stage V' transposed (branchless): vt[c][k] = Wh[b,k0+k,c] * rdenom[k0+k]
      int c = t & 63, kb = t >> 6;
#pragma unroll
      for (int r = 0; r < 8; ++r) {
        int k = kb * 8 + r;
        int gk = k0 + k;
        int gkc = (gk < N2) ? gk : (N2 - 1);
        float v = Wh[(b * N2 + gkc) * DD + c] * rdenom[b * N2 + gkc];
        if (gk >= N2) v = 0.f;
        vt[c * 40 + k] = (_Float16)v;
      }
    }
    __syncthreads();
    v16h af0 = load_fragA(&pt[(wrow +      (lane & 15)) * 40], lane);
    v16h af1 = load_fragA(&pt[(wrow + 16 + (lane & 15)) * 40], lane);
    v16h bf0 = load_fragB(&vt[(wcol +      (lane & 15)) * 40], lane);
    v16h bf1 = load_fragB(&vt[(wcol + 16 + (lane & 15)) * 40], lane);
    acc00 = __builtin_amdgcn_wmma_f32_16x16x32_f16(false, af0, false, bf0, (short)0, acc00, false, false);
    acc01 = __builtin_amdgcn_wmma_f32_16x16x32_f16(false, af0, false, bf1, (short)0, acc01, false, false);
    acc10 = __builtin_amdgcn_wmma_f32_16x16x32_f16(false, af1, false, bf0, (short)0, acc10, false, false);
    acc11 = __builtin_amdgcn_wmma_f32_16x16x32_f16(false, af1, false, bf1, (short)0, acc11, false, false);
  }

  // Store with ELU + concat([:307],[307:]) on channel axis. C layout: VGPR g ->
  // M = g (+8 for lanes>=16), N = lane&15.
  int cM = (lane >> 4) * 8, cN = lane & 15;
#pragma unroll
  for (int sub = 0; sub < 2; ++sub) {
    v8f a0 = sub ? acc10 : acc00;
    v8f a1 = sub ? acc11 : acc01;
#pragma unroll
    for (int g = 0; g < 8; ++g) {
      int n = row0 + wrow + sub * 16 + cM + g;
      if (n >= N2) continue;
      float v0 = a0[g], v1 = a1[g];
      v0 = v0 > 0.f ? v0 : (__expf(v0) - 1.f);
      v1 = v1 > 0.f ? v1 : (__expf(v1) - 1.f);
      int oi, oc;
      if (n < NN) { oi = n;      oc = wcol + cN; }
      else        { oi = n - NN; oc = 64 + wcol + cN; }
      out[(b * NN + oi) * 128 + oc]      = v0;
      out[(b * NN + oi) * 128 + oc + 16] = v1;
    }
  }
}

extern "C" void kernel_launch(void* const* d_in, const int* in_sizes, int n_in,
                              void* d_out, int out_size, void* d_ws, size_t ws_size,
                              hipStream_t stream) {
  const float* h   = (const float*)d_in[0];
  const float* ht  = (const float*)d_in[1];
  const float* W   = (const float*)d_in[2];
  const float* a   = (const float*)d_in[3];
  const float* adj = (const float*)d_in[4];
  float* out = (float*)d_out;

  // workspace layout (floats): Wh | wh1 | wh2 | cmax | rdenom  (~10.7 MB)
  float* Wh     = (float*)d_ws;
  float* wh1    = Wh   + (size_t)NB * N2 * DD;
  float* wh2    = wh1  + (size_t)NB * N2;
  float* cmax   = wh2  + (size_t)NB * N2;
  float* rdenom = cmax + (size_t)NB * N2;

  k_wh_gemm<<<dim3(NB * RT1), dim3(256), 0, stream>>>(h, ht, W, Wh);
  k_wh12<<<dim3((NB * N2 + 255) / 256), dim3(256), 0, stream>>>(Wh, a, wh1, wh2);
  k_colstats<<<dim3(NB * 3), dim3(256), 0, stream>>>(adj, wh1, wh2, cmax, rdenom);
  k_attn<<<dim3(NB * RT4), dim3(256), 0, stream>>>(adj, Wh, wh1, wh2, cmax, rdenom, out);
}
